// CSPLayer_25280177504324
// MI455X (gfx1250) — matrix operand order repaired and checked by reading, not compiled, not run on hardware
//
#include <hip/hip_runtime.h>
#include <hip/hip_bf16.h>

typedef __attribute__((ext_vector_type(16))) _Float16 v16h;
typedef __attribute__((ext_vector_type(8)))  _Float16 v8h;
typedef __attribute__((ext_vector_type(8)))  float    v8f;

#define HID    256       // hidden size
#define MT     32        // edges/nodes per block (2 M-tiles of 16)
#define KT_E1  29        // ceil(905/32) k-tiles for edge MLP1 (padded to 928)
#define KT_256 8         // 256/32
#define KT_512 16        // 512/32
#define ASTR   936       // LDS row stride (halves) for 905-wide edge input (bank stagger)
#define E1STR  264       // LDS row stride for 256-wide intermediate
#define NASTR  520       // LDS row stride for 512-wide node input

static __device__ __forceinline__ float silu_f(float x) {
    return x / (1.0f + __expf(-x));
}

static __device__ __forceinline__ v8f wmma16(v16h a, v16h b, v8f c) {
    // D = A(16x32 f16) * B(32x16 f16) + C(16x16 f32)
    return __builtin_amdgcn_wmma_f32_16x16x32_f16(false, a, false, b, (short)0, c, false, false);
}

// A fragment: per ISA 7.12.2 16-bit A 16x32 layout.
// lane r=lane&15 -> row M=r ; half=lane>>4 ; two contiguous 8-half runs per lane:
// [kt*32+half*8 .. +7] and [kt*32+16+half*8 .. +7]
static __device__ __forceinline__ v16h load_a_frag(const _Float16* row, int kt, int half) {
    v8h lo = *(const v8h*)(row + kt * 32 + half * 8);
    v8h hi = *(const v8h*)(row + kt * 32 + 16 + half * 8);
    v16h a;
#pragma unroll
    for (int i = 0; i < 8; ++i) { a[i] = lo[i]; a[8 + i] = hi[i]; }
    return a;
}

// B fragment: weights pre-swizzled so each lane's 16 halves are contiguous (32B load).
static __device__ __forceinline__ v16h load_b_frag(const _Float16* __restrict__ w,
                                                   int kt, int nt, int KT, int lane) {
    return *(const v16h*)(w + ((size_t)((nt * KT + kt) * 32 + lane)) * 16);
}

// ---------------- LayerNorm: h[N,256] f32 -> hln16[N,256] f16 -----------------
__global__ __launch_bounds__(256) void ln_kernel(const float* __restrict__ h,
                                                 const float* __restrict__ gamma,
                                                 const float* __restrict__ beta,
                                                 _Float16* __restrict__ hln16, int N_) {
    int wave = threadIdx.x >> 5, lane = threadIdx.x & 31;
    int row = blockIdx.x * 8 + wave;
    if (row >= N_) return;
    const float* x = h + (size_t)row * HID;
    float vals[8];
    float s = 0.f;
#pragma unroll
    for (int i = 0; i < 8; ++i) { vals[i] = x[lane + 32 * i]; s += vals[i]; }
#pragma unroll
    for (int off = 16; off >= 1; off >>= 1) s += __shfl_xor(s, off, 32);
    float mu = s * (1.0f / HID);
    float vs = 0.f;
#pragma unroll
    for (int i = 0; i < 8; ++i) { float d = vals[i] - mu; vs += d * d; }
#pragma unroll
    for (int off = 16; off >= 1; off >>= 1) vs += __shfl_xor(vs, off, 32);
    float rs = rsqrtf(vs * (1.0f / HID) + 1e-5f);
    _Float16* o = hln16 + (size_t)row * HID;
#pragma unroll
    for (int i = 0; i < 8; ++i) {
        int c = lane + 32 * i;
        o[c] = (_Float16)((vals[i] - mu) * rs * gamma[c] + beta[c]);
    }
}

// ---------------- lat_ip[g,i,k] = sum_j L[g,i,j]*L[g,k,j] -> f16 --------------
__global__ void latip_kernel(const float* __restrict__ lat, _Float16* __restrict__ latip16, int G_) {
    int t = blockIdx.x * 128 + threadIdx.x;
    if (t >= G_ * 9) return;
    int g = t / 9, ik = t % 9, i = ik / 3, k = ik % 3;
    const float* L = lat + (size_t)g * 9;
    float s = L[i * 3 + 0] * L[k * 3 + 0] + L[i * 3 + 1] * L[k * 3 + 1] + L[i * 3 + 2] * L[k * 3 + 2];
    latip16[t] = (_Float16)s;
}

// ------------- Weight swizzle: f32 [K,256] row-major -> f16 B-fragments -------
// B layout: lane half=lane>>4, matching A fragment order:
// k = kt*32 + (v>>2)*16 + half*8 + (v&3)*2 (+0/1), column n = nt*16 + (lane&15).
__global__ void swz_kernel(const float* __restrict__ src, _Float16* __restrict__ dst,
                           int K, int KT) {
    int t = blockIdx.x * 256 + threadIdx.x;
    int total = KT * 16 * 32 * 8;     // N is always 256 -> 16 n-tiles
    if (t >= total) return;
    int v = t & 7;
    int lane = (t >> 3) & 31;
    int kt = (t >> 8) % KT;
    int nt = (t >> 8) / KT;
    int half = lane >> 4, r = lane & 15;
    int n = nt * 16 + r;
    int k0 = kt * 32 + (v >> 2) * 16 + half * 8 + (v & 3) * 2;
    float x0 = (k0 < K) ? src[(size_t)k0 * 256 + n] : 0.f;
    float x1 = (k0 + 1 < K) ? src[(size_t)(k0 + 1) * 256 + n] : 0.f;
    size_t base = ((size_t)((nt * KT + kt) * 32 + lane)) * 16 + (size_t)v * 2;
    dst[base]     = (_Float16)x0;
    dst[base + 1] = (_Float16)x1;
}

// ---------------- degree count: cnt[seg[e]] += 1 ------------------------------
__global__ void count_kernel(const int* __restrict__ seg, float* __restrict__ cnt, int E_) {
    int e = blockIdx.x * 256 + threadIdx.x;
    if (e < E_) atomicAdd(&cnt[seg[e]], 1.0f);
}

// ---------------- Edge MLP + scatter-add (32 edges / block) -------------------
__global__ __launch_bounds__(256) void edge_mlp_kernel(
    const _Float16* __restrict__ hln16, const _Float16* __restrict__ latip16,
    const int* __restrict__ eidx, const int* __restrict__ e2g,
    const float* __restrict__ fd,
    const _Float16* __restrict__ eW1s, const float* __restrict__ eb1,
    const _Float16* __restrict__ eW2s, const float* __restrict__ eb2,
    float* __restrict__ agg, int E_) {
    __shared__ _Float16 Abuf[MT * ASTR];   // 59.9 KB; e1 (MT*E1STR, 16.9 KB) overlaid after GEMM1
    __shared__ int si[MT], sj[MT], sg[MT];
    int tid = threadIdx.x;
    int e0 = blockIdx.x * MT;

    if (tid < MT) {
        int e = min(e0 + tid, E_ - 1);
        si[tid] = eidx[e];
        sj[tid] = eidx[E_ + e];
        sg[tid] = e2g[e];
    }
    unsigned int* A32 = (unsigned int*)Abuf;
    for (int i = tid; i < MT * ASTR / 2; i += 256) A32[i] = 0u;
    __syncthreads();

    // hi | hj (dword copies from f16 table)
    for (int i = tid; i < MT * 128; i += 256) {
        int m = i >> 7, c = i & 127;
        unsigned int* row = (unsigned int*)(Abuf + m * ASTR);
        row[c]       = ((const unsigned int*)(hln16 + (size_t)si[m] * HID))[c];
        row[128 + c] = ((const unsigned int*)(hln16 + (size_t)sj[m] * HID))[c];
    }
    // lat_ip (9 cols at 512)
    for (int i = tid; i < MT * 9; i += 256) {
        int m = i / 9, j = i % 9;
        Abuf[m * ASTR + 512 + j] = latip16[sg[m] * 9 + j];
    }
    // sinusoids: cols [521,713)=sin, [713,905)=cos, space-major then freq
    for (int i = tid; i < MT * 192; i += 256) {
        int m = i / 192, r = i % 192;
        int coord = r >> 6, f = r & 63;
        int ee = min(e0 + m, E_ - 1);
        float x = fd[(size_t)ee * 3 + coord];
        float arg = x * (6.283185307179586f * (float)f);
        float sv, cv;
        __sincosf(arg, &sv, &cv);
        Abuf[m * ASTR + 521 + r] = (_Float16)sv;
        Abuf[m * ASTR + 713 + r] = (_Float16)cv;
    }
    __syncthreads();

    int wave = tid >> 5, lane = tid & 31;
    int half = lane >> 4, r = lane & 15;
    const _Float16* arow0 = Abuf + r * ASTR;
    const _Float16* arow1 = Abuf + (16 + r) * ASTR;
    int n0 = wave * 16 + r, n1 = (wave + 8) * 16 + r;

    // GEMM1: [32,928] x [928,256]; wave owns n-tiles {wave, wave+8}, m-tiles {0,1}
    v8f c00 = {}, c01 = {}, c10 = {}, c11 = {};
    for (int kt = 0; kt < KT_E1; ++kt) {
        v16h a0 = load_a_frag(arow0, kt, half);
        v16h a1 = load_a_frag(arow1, kt, half);
        v16h b0 = load_b_frag(eW1s, kt, wave, KT_E1, lane);
        v16h b1 = load_b_frag(eW1s, kt, wave + 8, KT_E1, lane);
        c00 = wmma16(a0, b0, c00);
        c01 = wmma16(a0, b1, c01);
        c10 = wmma16(a1, b0, c10);
        c11 = wmma16(a1, b1, c11);
    }
    __syncthreads();                 // Abuf dead; reuse as e1
    _Float16* e1 = Abuf;
    {
        float bb0 = eb1[n0], bb1 = eb1[n1];
#pragma unroll
        for (int v = 0; v < 8; ++v) {
            int m0 = half * 8 + v, m1 = 16 + half * 8 + v;
            e1[m0 * E1STR + n0] = (_Float16)silu_f(c00[v] + bb0);
            e1[m0 * E1STR + n1] = (_Float16)silu_f(c01[v] + bb1);
            e1[m1 * E1STR + n0] = (_Float16)silu_f(c10[v] + bb0);
            e1[m1 * E1STR + n1] = (_Float16)silu_f(c11[v] + bb1);
        }
    }
    __syncthreads();

    // GEMM2: [32,256] x [256,256]
    v8f d00 = {}, d01 = {}, d10 = {}, d11 = {};
    const _Float16* erow0 = e1 + r * E1STR;
    const _Float16* erow1 = e1 + (16 + r) * E1STR;
    for (int kt = 0; kt < KT_256; ++kt) {
        v16h a0 = load_a_frag(erow0, kt, half);
        v16h a1 = load_a_frag(erow1, kt, half);
        v16h b0 = load_b_frag(eW2s, kt, wave, KT_256, lane);
        v16h b1 = load_b_frag(eW2s, kt, wave + 8, KT_256, lane);
        d00 = wmma16(a0, b0, d00);
        d01 = wmma16(a0, b1, d01);
        d10 = wmma16(a1, b0, d10);
        d11 = wmma16(a1, b1, d11);
    }
    {
        float bb0 = eb2[n0], bb1 = eb2[n1];
#pragma unroll
        for (int v = 0; v < 8; ++v) {
            int m0 = half * 8 + v, m1 = 16 + half * 8 + v;
            if (e0 + m0 < E_) {
                float* dr = agg + (size_t)si[m0] * HID;
                atomicAdd(dr + n0, silu_f(d00[v] + bb0));
                atomicAdd(dr + n1, silu_f(d01[v] + bb1));
            }
            if (e0 + m1 < E_) {
                float* dr = agg + (size_t)si[m1] * HID;
                atomicAdd(dr + n0, silu_f(d10[v] + bb0));
                atomicAdd(dr + n1, silu_f(d11[v] + bb1));
            }
        }
    }
}

// ---------------- Node MLP + residual (32 nodes / block) ----------------------
__global__ __launch_bounds__(256) void node_mlp_kernel(
    const float* __restrict__ h_in, const _Float16* __restrict__ hln16,
    const float* __restrict__ cnt,
    const _Float16* __restrict__ nW1s, const float* __restrict__ nb1,
    const _Float16* __restrict__ nW2s, const float* __restrict__ nb2,
    float* __restrict__ out /* holds agg_sum on entry */, int N_) {
    __shared__ _Float16 Abuf[MT * NASTR];  // 33.3 KB; e1 (16.9 KB) overlaid after GEMM1
    int tid = threadIdx.x;
    int nb0 = blockIdx.x * MT;

    // cols [0,256)=h_ln, [256,512)=agg_sum/max(cnt,1)
    for (int i = tid; i < MT * 128; i += 256) {
        int m = i >> 7, c = i & 127;
        int rr = min(nb0 + m, N_ - 1);
        ((unsigned int*)(Abuf + m * NASTR))[c] = ((const unsigned int*)(hln16 + (size_t)rr * HID))[c];
    }
    for (int i = tid; i < MT * HID; i += 256) {
        int m = i >> 8, c = i & 255;
        int rr = min(nb0 + m, N_ - 1);
        float inv = 1.0f / fmaxf(cnt[rr], 1.0f);
        Abuf[m * NASTR + HID + c] = (_Float16)(out[(size_t)rr * HID + c] * inv);
    }
    for (int i = tid; i < MT * 8; i += 256) {
        int m = i >> 3, j = i & 7;
        Abuf[m * NASTR + 512 + j] = (_Float16)0.f;
    }
    __syncthreads();

    int wave = tid >> 5, lane = tid & 31;
    int half = lane >> 4, r = lane & 15;
    const _Float16* arow0 = Abuf + r * NASTR;
    const _Float16* arow1 = Abuf + (16 + r) * NASTR;
    int n0 = wave * 16 + r, n1 = (wave + 8) * 16 + r;

    v8f c00 = {}, c01 = {}, c10 = {}, c11 = {};
    for (int kt = 0; kt < KT_512; ++kt) {
        v16h a0 = load_a_frag(arow0, kt, half);
        v16h a1 = load_a_frag(arow1, kt, half);
        v16h b0 = load_b_frag(nW1s, kt, wave, KT_512, lane);
        v16h b1 = load_b_frag(nW1s, kt, wave + 8, KT_512, lane);
        c00 = wmma16(a0, b0, c00);
        c01 = wmma16(a0, b1, c01);
        c10 = wmma16(a1, b0, c10);
        c11 = wmma16(a1, b1, c11);
    }
    __syncthreads();                 // Abuf dead; reuse as e1
    _Float16* e1 = Abuf;
    {
        float bb0 = nb1[n0], bb1 = nb1[n1];
#pragma unroll
        for (int v = 0; v < 8; ++v) {
            int m0 = half * 8 + v, m1 = 16 + half * 8 + v;
            e1[m0 * E1STR + n0] = (_Float16)silu_f(c00[v] + bb0);
            e1[m0 * E1STR + n1] = (_Float16)silu_f(c01[v] + bb1);
            e1[m1 * E1STR + n0] = (_Float16)silu_f(c10[v] + bb0);
            e1[m1 * E1STR + n1] = (_Float16)silu_f(c11[v] + bb1);
        }
    }
    __syncthreads();

    v8f d00 = {}, d01 = {}, d10 = {}, d11 = {};
    const _Float16* erow0 = e1 + r * E1STR;
    const _Float16* erow1 = e1 + (16 + r) * E1STR;
    for (int kt = 0; kt < KT_256; ++kt) {
        v16h a0 = load_a_frag(erow0, kt, half);
        v16h a1 = load_a_frag(erow1, kt, half);
        v16h b0 = load_b_frag(nW2s, kt, wave, KT_256, lane);
        v16h b1 = load_b_frag(nW2s, kt, wave + 8, KT_256, lane);
        d00 = wmma16(a0, b0, d00);
        d01 = wmma16(a0, b1, d01);
        d10 = wmma16(a1, b0, d10);
        d11 = wmma16(a1, b1, d11);
    }
    {
        float bb0 = nb2[n0], bb1 = nb2[n1];
#pragma unroll
        for (int v = 0; v < 8; ++v) {
            int m0 = half * 8 + v, m1 = 16 + half * 8 + v;
            int row0 = nb0 + m0, row1 = nb0 + m1;
            if (row0 < N_) {
                const float* hr = h_in + (size_t)row0 * HID;
                float* orow = out + (size_t)row0 * HID;
                orow[n0] = hr[n0] + silu_f(d00[v] + bb0);
                orow[n1] = hr[n1] + silu_f(d01[v] + bb1);
            }
            if (row1 < N_) {
                const float* hr = h_in + (size_t)row1 * HID;
                float* orow = out + (size_t)row1 * HID;
                orow[n0] = hr[n0] + silu_f(d10[v] + bb0);
                orow[n1] = hr[n1] + silu_f(d11[v] + bb1);
            }
        }
    }
}

extern "C" void kernel_launch(void* const* d_in, const int* in_sizes, int n_in,
                              void* d_out, int out_size, void* d_ws, size_t ws_size,
                              hipStream_t stream) {
    const float* h         = (const float*)d_in[0];
    const float* lattices  = (const float*)d_in[2];
    const int*   eidx      = (const int*)d_in[3];
    const int*   e2g       = (const int*)d_in[4];
    const float* frac_diff = (const float*)d_in[5];
    const float* ln_gamma  = (const float*)d_in[6];
    const float* ln_beta   = (const float*)d_in[7];
    const float* eW1 = (const float*)d_in[8];
    const float* eb1 = (const float*)d_in[9];
    const float* eW2 = (const float*)d_in[10];
    const float* eb2 = (const float*)d_in[11];
    const float* nW1 = (const float*)d_in[12];
    const float* nb1 = (const float*)d_in[13];
    const float* nW2 = (const float*)d_in[14];
    const float* nb2 = (const float*)d_in[15];
    float* out = (float*)d_out;

    const int N_ = in_sizes[0] / HID;
    const int G_ = in_sizes[2] / 9;
    const int E_ = in_sizes[4];

    // workspace carve-up (256B aligned)
    char* ws = (char*)d_ws;
    size_t off = 0;
    auto carve = [&](size_t bytes) -> char* {
        char* p = ws + off;
        off += (bytes + 255) & ~(size_t)255;
        return p;
    };
    _Float16* hln16   = (_Float16*)carve((size_t)N_ * HID * 2);
    _Float16* latip16 = (_Float16*)carve((size_t)G_ * 9 * 2);
    _Float16* eW1s    = (_Float16*)carve((size_t)KT_E1 * 16 * 1024);
    _Float16* eW2s    = (_Float16*)carve((size_t)KT_256 * 16 * 1024);
    _Float16* nW1s    = (_Float16*)carve((size_t)KT_512 * 16 * 1024);
    _Float16* nW2s    = (_Float16*)carve((size_t)KT_256 * 16 * 1024);
    float*    cnt     = (float*)carve((size_t)N_ * 4);
    (void)ws_size;

    // 0) zero the scatter accumulator (d_out doubles as agg_sum) and counts
    hipMemsetAsync(d_out, 0, (size_t)N_ * HID * sizeof(float), stream);
    hipMemsetAsync(cnt, 0, (size_t)N_ * sizeof(float), stream);

    // 1) layernorm -> f16
    ln_kernel<<<(N_ + 7) / 8, 256, 0, stream>>>(h, ln_gamma, ln_beta, hln16, N_);
    // 2) L @ L^T per graph
    latip_kernel<<<(G_ * 9 + 127) / 128, 128, 0, stream>>>(lattices, latip16, G_);
    // 3) weight swizzles into WMMA fragment layout
    swz_kernel<<<(KT_E1 * 16 * 32 * 8 + 255) / 256, 256, 0, stream>>>(eW1, eW1s, 905, KT_E1);
    swz_kernel<<<(KT_256 * 16 * 32 * 8 + 255) / 256, 256, 0, stream>>>(eW2, eW2s, 256, KT_256);
    swz_kernel<<<(KT_512 * 16 * 32 * 8 + 255) / 256, 256, 0, stream>>>(nW1, nW1s, 512, KT_512);
    swz_kernel<<<(KT_256 * 16 * 32 * 8 + 255) / 256, 256, 0, stream>>>(nW2, nW2s, 256, KT_256);
    // 4) degree counts
    count_kernel<<<(E_ + 255) / 256, 256, 0, stream>>>(eidx, cnt, E_);
    // 5) edge MLP + scatter-add into d_out
    edge_mlp_kernel<<<(E_ + MT - 1) / MT, 256, 0, stream>>>(hln16, latip16, eidx, e2g, frac_diff,
                                                            eW1s, eb1, eW2s, eb2, out, E_);
    // 6) node MLP + residual, overwrites d_out in place (block-local rows only)
    node_mlp_kernel<<<(N_ + MT - 1) / MT, 256, 0, stream>>>(h, hln16, cnt, nW1s, nb1, nW2s, nb2,
                                                            out, N_);
}
